// GNN_46789373722956
// MI455X (gfx1250) — matrix operand order repaired and checked by reading, not compiled, not run on hardware
//
#include <hip/hip_runtime.h>
#include <hip/hip_bf16.h>
#include <math.h>

typedef __attribute__((ext_vector_type(16))) _Float16 v16h;
typedef __attribute__((ext_vector_type(8)))  _Float16 v8h;
typedef __attribute__((ext_vector_type(8)))  float    v8f;

#define IN_CH        128
#define OUTW         128   // HEADS * OUT_NEURONS
#define HEADS        4
#define OUT_NEURONS  32
#define OUT_CH       16
#define NEG_SLOPE    0.2f

// ---------------------------------------------------------------------------
// Kernel 1: convert x (f32) -> f16
__global__ void cvt_x_f16(const float* __restrict__ x, _Float16* __restrict__ x16, int n) {
    int i = blockIdx.x * blockDim.x + threadIdx.x;
    if (i < n) x16[i] = (_Float16)x[i];
}

// Kernel 2: convert weight (128x128 f32, [k][n]) -> transposed f16 [n][k]
__global__ void cvt_w_f16(const float* __restrict__ w, _Float16* __restrict__ wt) {
    int i = blockIdx.x * blockDim.x + threadIdx.x;   // i = k*128 + n
    if (i < IN_CH * OUTW) {
        int k = i >> 7, n = i & 127;
        wt[n * IN_CH + k] = (_Float16)w[i];
    }
}

// Kernel 3: init scratch: m = -inf, denom = 0, agg = 0
__global__ void init_ws(float* __restrict__ m, float* __restrict__ denom,
                        float* __restrict__ agg, int nN) {
    int p = blockIdx.x * blockDim.x + threadIdx.x;
    int n4 = 4 * nN;
    if (p < n4)            m[p]      = -__builtin_inff();
    else if (p < 2 * n4)   denom[p - n4] = 0.f;
    else if (p < 2 * n4 + 128 * nN) agg[p - 2 * n4] = 0.f;
}

// ---------------------------------------------------------------------------
// Kernel 4: h = x @ W via WMMA f32 <- f16xf16, K=128 (4 chained 16x16x32).
// Block = 256 threads = 8 waves; block handles one 16-row tile, wave w
// handles column tile w (16 cols). EXEC is all-ones (no divergence) as WMMA
// requires.
__global__ __launch_bounds__(256) void gemm_wmma(const _Float16* __restrict__ x16,
                                                 const _Float16* __restrict__ wt,
                                                 float* __restrict__ h) {
    const int wave = threadIdx.x >> 5;     // 0..7 -> column tile
    const int lane = threadIdx.x & 31;
    const int half = lane >> 4;            // 0/1
    const int l15  = lane & 15;
    const int row0 = blockIdx.x * 16;

    // A fragment source: row (row0+l15), K runs [half*8 .. half*8+7] and +16
    const _Float16* arow = x16 + (size_t)(row0 + l15) * IN_CH + half * 8;
    // B fragment source: column (wave*16 + l15) of W == row of Wt, K runs [half*8 .. half*8+7] and +16
    const _Float16* bcol = wt + (size_t)(wave * 16 + l15) * IN_CH + half * 8;

    v8f acc = {};
#pragma unroll
    for (int kc = 0; kc < 4; ++kc) {
        v8h alo = *(const v8h*)(arow + kc * 32);        // K = kc*32 + half*8 + 0..7
        v8h ahi = *(const v8h*)(arow + kc * 32 + 16);   // K = kc*32 + 16 + half*8 + 0..7
        v16h a;
#pragma unroll
        for (int i = 0; i < 8; ++i) { a[i] = alo[i]; a[i + 8] = ahi[i]; }
        v8h blo = *(const v8h*)(bcol + kc * 32); v8h bhi = *(const v8h*)(bcol + kc * 32 + 16); v16h b; for (int i = 0; i < 8; ++i) { b[i] = blo[i]; b[i + 8] = bhi[i]; }  // K = kc*32 + half*8 + 0..7, then +16
        acc = __builtin_amdgcn_wmma_f32_16x16x32_f16(
            /*neg_a=*/false, a, /*neg_b=*/false, b,
            /*c_mod=*/(short)0, acc, /*reuse_a=*/false, /*reuse_b=*/false);
    }
    // D layout: element r -> row M = r + 8*half, col N = l15
    float* hout = h + (size_t)(row0 + half * 8) * OUTW + wave * 16 + l15;
#pragma unroll
    for (int r = 0; r < 8; ++r) hout[(size_t)r * OUTW] = acc[r];
}

// ---------------------------------------------------------------------------
// Kernel 5: per-(node, head) attention reductions
// ai[n][hd] = sum_c h[n][hd*32+c] * att[hd*64 + c]       (x_i part)
// aj[n][hd] = sum_c h[n][hd*32+c] * att[hd*64 + 32 + c]  (x_j part)
__global__ void att_reduce(const float* __restrict__ h, const float* __restrict__ att,
                           float* __restrict__ ai, float* __restrict__ aj, int nN) {
    int g = blockIdx.x * blockDim.x + threadIdx.x;
    if (g >= nN * HEADS) return;
    int n = g >> 2, hd = g & 3;
    const float* hp = h + (size_t)n * OUTW + hd * OUT_NEURONS;
    const float* ap = att + hd * (2 * OUT_NEURONS);
    float si = 0.f, sj = 0.f;
#pragma unroll
    for (int c = 0; c < OUT_NEURONS; ++c) {
        float v = hp[c];
        si += v * ap[c];
        sj += v * ap[OUT_NEURONS + c];
    }
    ai[g] = si;
    aj[g] = sj;
}

// float atomic max via signed/unsigned integer ordering trick (init = -inf)
__device__ inline void atomicMaxF(float* addr, float val) {
    int iv = __float_as_int(val);
    if (iv >= 0) atomicMax((int*)addr, iv);
    else         atomicMin((unsigned int*)addr, (unsigned int)iv);
}

// Kernel 6: edge logits + leaky relu + segment max over dst
__global__ void edge_logit(const int* __restrict__ src, const int* __restrict__ dst,
                           const float* __restrict__ ai, const float* __restrict__ aj,
                           float* __restrict__ alphaE, float* __restrict__ m, int E) {
    int e = blockIdx.x * blockDim.x + threadIdx.x;
    if (e >= E) return;
    int s = src[e], d = dst[e];
#pragma unroll
    for (int hd = 0; hd < HEADS; ++hd) {
        float a = ai[d * HEADS + hd] + aj[s * HEADS + hd];
        a = (a >= 0.f) ? a : NEG_SLOPE * a;          // leaky_relu
        alphaE[(size_t)e * HEADS + hd] = a;
        atomicMaxF(&m[d * HEADS + hd], a);
    }
}

// Kernel 7: e = exp(alpha - m[dst]); denom[dst] += e
__global__ void edge_exp(const int* __restrict__ dst, const float* __restrict__ m,
                         float* __restrict__ alphaE, float* __restrict__ denom, int E) {
    int e = blockIdx.x * blockDim.x + threadIdx.x;
    if (e >= E) return;
    int d = dst[e];
#pragma unroll
    for (int hd = 0; hd < HEADS; ++hd) {
        float ex = __expf(alphaE[(size_t)e * HEADS + hd] - m[d * HEADS + hd]);
        alphaE[(size_t)e * HEADS + hd] = ex;
        atomicAdd(&denom[d * HEADS + hd], ex);
    }
}

// Kernel 8: weighted scatter: agg[dst][c] += h[src][c] * (e / (denom[dst]+eps))
// One thread per (edge, channel). h/agg are L2-resident (25.6 MB each).
__global__ void edge_scatter(const int* __restrict__ src, const int* __restrict__ dst,
                             const float* __restrict__ h, const float* __restrict__ alphaE,
                             const float* __restrict__ denom, float* __restrict__ agg,
                             int total) {
    int g = blockIdx.x * blockDim.x + threadIdx.x;
    if (g >= total) return;
    int e = g >> 7;          // edge
    int c = g & 127;         // channel 0..127
    int hd = c >> 5;
    int s = src[e], d = dst[e];
    float w = alphaE[(size_t)e * HEADS + hd] / (denom[d * HEADS + hd] + 1e-16f);
    atomicAdd(&agg[(size_t)d * OUTW + c], h[(size_t)s * OUTW + c] * w);
}

// Kernel 9: variational head. One thread per node.
__device__ inline float softplusf(float v) {
    return (v > 20.f) ? v : log1pf(expf(v));
}
__global__ void finalize(const float* __restrict__ agg, const float* __restrict__ bias,
                         float* __restrict__ out_final, float* __restrict__ out_ixz,
                         float* __restrict__ out_skl, float* __restrict__ out_mean,
                         float* __restrict__ out_std, int nN) {
    int n = blockIdx.x * blockDim.x + threadIdx.x;
    if (n >= nN) return;
    float kls = 0.f;
#pragma unroll
    for (int hd = 0; hd < HEADS; ++hd) {
        float kl = 0.f;
#pragma unroll
        for (int j = 0; j < OUT_CH; ++j) {
            float mean = agg[(size_t)n * OUTW + hd * OUT_NEURONS + j]
                       + bias[hd * OUT_NEURONS + j];
            float raw  = agg[(size_t)n * OUTW + hd * OUT_NEURONS + OUT_CH + j]
                       + bias[hd * OUT_NEURONS + OUT_CH + j] - 5.0f;
            float sd = softplusf(raw) + 1e-10f;
            out_final[(size_t)n * (HEADS * OUT_CH) + hd * OUT_CH + j] = mean;
            out_mean[((size_t)n * HEADS + hd) * OUT_CH + j] = mean;
            out_std [((size_t)n * HEADS + hd) * OUT_CH + j] = sd;
            kl += -logf(sd) + 0.5f * (sd * sd + mean * mean) - 0.5f;
        }
        kls += kl;
    }
    out_ixz[n] = kls * (1.0f / HEADS);
    if (n == 0) out_skl[0] = 0.f;
}

// ---------------------------------------------------------------------------
extern "C" void kernel_launch(void* const* d_in, const int* in_sizes, int n_in,
                              void* d_out, int out_size, void* d_ws, size_t ws_size,
                              hipStream_t stream) {
    const float* x    = (const float*)d_in[0];
    const int*   ei   = (const int*)  d_in[1];
    const float* w    = (const float*)d_in[2];
    const float* att  = (const float*)d_in[3];
    const float* bias = (const float*)d_in[4];

    const int N = in_sizes[0] / IN_CH;   // 50000
    const int E = in_sizes[1] / 2;       // 650000
    const int* src = ei;
    const int* dst = ei + E;

    // workspace layout (all offsets 256B-aligned for these sizes)
    char* ws = (char*)d_ws;
    size_t o = 0;
    _Float16* x16 = (_Float16*)(ws + o); o += (size_t)N * IN_CH * sizeof(_Float16);
    _Float16* wt  = (_Float16*)(ws + o); o += (size_t)IN_CH * OUTW * sizeof(_Float16);
    float* h      = (float*)(ws + o);    o += (size_t)N * OUTW * sizeof(float);
    float* ai     = (float*)(ws + o);    o += (size_t)N * HEADS * sizeof(float);
    float* aj     = (float*)(ws + o);    o += (size_t)N * HEADS * sizeof(float);
    float* m      = (float*)(ws + o);    o += (size_t)N * HEADS * sizeof(float);
    float* denom  = (float*)(ws + o);    o += (size_t)N * HEADS * sizeof(float);
    float* alphaE = (float*)(ws + o);    o += (size_t)E * HEADS * sizeof(float);
    float* agg    = (float*)(ws + o);    o += (size_t)N * OUTW * sizeof(float);

    float* out_final = (float*)d_out;                       // N x 64
    float* out_ixz   = out_final + (size_t)N * HEADS * OUT_CH;
    float* out_skl   = out_ixz + N;                         // scalar
    float* out_mean  = out_skl + 1;                         // N*4 x 16
    float* out_std   = out_mean + (size_t)N * HEADS * OUT_CH;

    const int T = 256;
    // 1) conversions
    cvt_x_f16<<<(N * IN_CH + T - 1) / T, T, 0, stream>>>(x, x16, N * IN_CH);
    cvt_w_f16<<<(IN_CH * OUTW + T - 1) / T, T, 0, stream>>>(w, wt);
    // 2) init m/denom/agg
    {
        int tot = 4 * N + 4 * N + 128 * N;
        init_ws<<<(tot + T - 1) / T, T, 0, stream>>>(m, denom, agg, N);
    }
    // 3) GEMM: 16-row tile per block, 8 waves cover 128 output cols
    gemm_wmma<<<N / 16, 256, 0, stream>>>(x16, wt, h);
    // 4) per-node attention reductions
    att_reduce<<<(N * HEADS + T - 1) / T, T, 0, stream>>>(h, att, ai, aj, N);
    // 5) edge logits + segment max
    edge_logit<<<(E + T - 1) / T, T, 0, stream>>>(src, dst, ai, aj, alphaE, m, E);
    // 6) exp + segment sum
    edge_exp<<<(E + T - 1) / T, T, 0, stream>>>(dst, m, alphaE, denom, E);
    // 7) weighted scatter aggregation
    {
        int tot = E * OUTW;   // 83.2M threads, one atomic each
        edge_scatter<<<(tot + T - 1) / T, T, 0, stream>>>(src, dst, h, alphaE, denom, agg, tot);
    }
    // 8) variational head + outputs
    finalize<<<(N + T - 1) / T, T, 0, stream>>>(agg, bias, out_final, out_ixz,
                                                out_skl, out_mean, out_std, N);
}
